// RPEMultiHeadAttention_81054622810260
// MI455X (gfx1250) — compile-verified, matrix operand order checked by reading, not run on hardware
//
#include <hip/hip_runtime.h>

// ---------------------------------------------------------------------------
// RPE KNN multi-head attention for MI455X (gfx1250, wave32, WMMA).
// B=1, N=16384, C=256, H=8, D=32, K_NN=35.
//
// Pipeline:
//   1) wt_convert   : W[k][n] fp32 -> Wt[n][k] bf16 (5 weight matrices)
//   2) gemm_proj    : Y = X @ W + b via v_wmma_f32_16x16x32_bf16 (q,q1,k,v)
//   3) rpe_attn     : per-query block: p = rpe@Wp in LDS via WMMA, k/v
//                     neighbor gather via global_load_async_to_lds_b128
//                     (ASYNCcnt path), scores, softmax, AV.  bp cancels in
//                     softmax (constant over the softmax axis) -> omitted.
// ---------------------------------------------------------------------------

typedef __bf16 bf16;
typedef __attribute__((ext_vector_type(16))) __bf16 v16bf;
typedef __attribute__((ext_vector_type(4)))  __bf16 bf16x4;
typedef __attribute__((ext_vector_type(8)))  float  v8f;
typedef __attribute__((ext_vector_type(4)))  float  f32x4;

#define NPTS   16384
#define CDIM   256
#define KNN    35
#define NHEAD  8
#define HD     32

union Frag { v16bf v; f32x4 h[2]; };

// Load a 16x16x32 A/B fragment: two contiguous 8-bf16 (16B) chunks per lane.
// ISA 7.12.2: 16-bit A 16x32: lane = row%16; elems 0..7 -> K = (lane/16)*8+j,
// elems 8..15 -> K = 16 + (lane/16)*8 + (j-8).
__device__ __forceinline__ v16bf load_frag(const bf16* p0, const bf16* p1) {
    Frag f;
    f.h[0] = *(const f32x4*)p0;
    f.h[1] = *(const f32x4*)p1;
    return f.v;
}

__device__ __forceinline__ v8f wmma_bf16(v16bf a, v16bf b, v8f c) {
    return __builtin_amdgcn_wmma_f32_16x16x32_bf16(false, a, false, b,
                                                   (short)0, c, false, false);
}

// Workgroup-relative LDS byte offset of a generic shared pointer.
// Flat->LDS aperture mapping truncates to addr[31:0] (ISA 10.2), which is the
// same LDS_BASE-relative offset GLOBAL_LOAD_ASYNC_TO_LDS expects in VDST.
__device__ __forceinline__ unsigned lds_off(const void* p) {
    return (unsigned)(uintptr_t)p;
}

// Async Global -> LDS copy of 16 bytes per lane (tracked by ASYNCcnt).
__device__ __forceinline__ void async_copy_b128(unsigned lds_byte, const void* gaddr) {
    asm volatile("global_load_async_to_lds_b128 %0, %1, off"
                 :: "v"(lds_byte), "v"((unsigned long long)(uintptr_t)gaddr)
                 : "memory");
}

__device__ __forceinline__ void wait_asynccnt0() {
    asm volatile("s_wait_asynccnt 0x0" ::: "memory");
}

// ---------------------------------------------------------------------------
// Kernel 1: weight transpose + fp32->bf16.  Wt[n*256+k] = bf16(W[k*256+n]).
// ---------------------------------------------------------------------------
__global__ __launch_bounds__(256) void wt_convert(const float* __restrict__ W,
                                                  bf16* __restrict__ Wt) {
    int i = blockIdx.x * 256 + threadIdx.x;   // 65536 elements
    int n = i >> 8, k = i & 255;
    Wt[n * 256 + k] = (bf16)W[k * 256 + n];
}

// ---------------------------------------------------------------------------
// Kernel 2: Y[M,256] = X[M,256] @ W + bias, bf16 WMMA, fp32 accumulate.
// Block tile 128(M) x 64(N), 8 waves; each wave: 16x64 strip (4 accumulators).
// grid = (M/128, 256/64)
// ---------------------------------------------------------------------------
__global__ __launch_bounds__(256) void gemm_proj(const float* __restrict__ X,
                                                 const bf16*  __restrict__ Wt,
                                                 const float* __restrict__ bias,
                                                 float*       __restrict__ Y) {
    __shared__ bf16 As[128 * 40];   // 128 rows x 32 K, stride 40 (pad: banks)
    __shared__ bf16 Bs[64 * 40];    // 64 cols  x 32 K, stride 40

    const int tid  = threadIdx.x;
    const int lane = tid & 31;
    const int wave = tid >> 5;
    const int m0   = blockIdx.x * 128;
    const int n0   = blockIdx.y * 64;
    const int mrow  = lane & 15;
    const int khalf = (lane >> 4) * 8;

    v8f acc[4] = {};

    for (int ks = 0; ks < 256; ks += 32) {
        // Stage X tile fp32 -> bf16 (1024 float4 loads across 256 threads)
        #pragma unroll
        for (int j = 0; j < 4; ++j) {
            int lin = tid * 4 + j;            // 0..1023
            int r   = lin >> 3;               // row 0..127
            int c4  = lin & 7;                // float4 within 32-wide K slab
            f32x4 f = *(const f32x4*)(X + (size_t)(m0 + r) * 256 + ks + c4 * 4);
            bf16x4 b;
            b[0] = (bf16)f[0]; b[1] = (bf16)f[1];
            b[2] = (bf16)f[2]; b[3] = (bf16)f[3];
            *(bf16x4*)(As + r * 40 + c4 * 4) = b;
        }
        // Stage W tile (already bf16, N-major -> contiguous 16B chunks)
        {
            int nr = tid >> 2, chunk = tid & 3;
            f32x4 w = *(const f32x4*)(Wt + (size_t)(n0 + nr) * 256 + ks + chunk * 8);
            *(f32x4*)(Bs + nr * 40 + chunk * 8) = w;
        }
        __syncthreads();

        const bf16* ab = As + (wave * 16 + mrow) * 40 + khalf;
        v16bf a = load_frag(ab, ab + 16);
        #pragma unroll
        for (int nt = 0; nt < 4; ++nt) {
            const bf16* bb = Bs + (nt * 16 + mrow) * 40 + khalf;
            v16bf b = load_frag(bb, bb + 16);
            acc[nt] = wmma_bf16(a, b, acc[nt]);
        }
        __syncthreads();
    }

    // D layout: lane col = N%16; VGPR r -> row r + (lane/16)*8
    #pragma unroll
    for (int nt = 0; nt < 4; ++nt) {
        int col = n0 + nt * 16 + mrow;
        float bv = bias[col];
        #pragma unroll
        for (int r = 0; r < 8; ++r) {
            int m = m0 + wave * 16 + r + (lane >> 4) * 8;
            Y[(size_t)m * 256 + col] = acc[nt][r] + bv;
        }
    }
}

// ---------------------------------------------------------------------------
// Kernel 3: per-query fused RPE attention.  One 256-thread block per query.
// LDS: rpe bf16 48x264 | p f32 48x264 | knb/vnb f32 35x260 | q,q1 | scores.
// ---------------------------------------------------------------------------
#define RPE_STR 264     // bf16 elems per row (pad)
#define P_STR   264     // f32 elems per row
#define KV_STR  260     // f32 elems per row
#define OFF_P   (48 * RPE_STR * 2)                       // 25344
#define OFF_KNB (OFF_P   + 48 * P_STR * 4)               // +50688
#define OFF_VNB (OFF_KNB + KNN * KV_STR * 4)             // +36400
#define OFF_Q   (OFF_VNB + KNN * KV_STR * 4)             // +36400
#define OFF_Q1  (OFF_Q + 256 * 4)
#define OFF_SC  (OFF_Q1 + 256 * 4)
#define SMEM_BYTES (OFF_SC + NHEAD * 64 * 4)             // 152928 B

__global__ __launch_bounds__(256) void rpe_attn(
    const float* __restrict__ rpe,  const int* __restrict__ knn_idx,
    const bf16*  __restrict__ Wpt,
    const float* __restrict__ qb,   const float* __restrict__ q1b,
    const float* __restrict__ kb,   const float* __restrict__ vb,
    float* __restrict__ out_h,      float* __restrict__ out_p) {

    extern __shared__ char smem[];
    bf16*  s_rpe = (bf16*)smem;
    float* s_p   = (float*)(smem + OFF_P);
    float* s_knb = (float*)(smem + OFF_KNB);
    float* s_vnb = (float*)(smem + OFF_VNB);
    float* s_q   = (float*)(smem + OFF_Q);
    float* s_q1  = (float*)(smem + OFF_Q1);
    float* s_sc  = (float*)(smem + OFF_SC);

    const int n    = blockIdx.x;
    const int tid  = threadIdx.x;
    const int lane = tid & 31;
    const int wave = tid >> 5;

    // -------- phase 0: async k/v gather, q/q1 row, rpe tile (fp32->bf16) ---
    // k/v neighbor gather: pure memory->LDS copy, no conversion needed ->
    // use the CDNA5 async path (ASYNCcnt), overlapping with rpe staging.
    for (int i = tid; i < KNN * 64; i += 256) {
        int kk = i >> 6, c4 = i & 63;
        int row = knn_idx[(size_t)n * KNN + kk];
        const float* gk = kb + (size_t)row * 256 + c4 * 4;
        const float* gv = vb + (size_t)row * 256 + c4 * 4;
        async_copy_b128(lds_off(s_knb + kk * KV_STR + c4 * 4), gk);
        async_copy_b128(lds_off(s_vnb + kk * KV_STR + c4 * 4), gv);
    }

    s_q[tid]  = qb[(size_t)n * 256 + tid];
    s_q1[tid] = q1b[(size_t)n * 256 + tid];

    // Warm L2/WGP$ for this wave's Wp^T column stream (B fragments, phase 1).
    __builtin_prefetch(Wpt + (size_t)(wave * 2) * 16 * 256 + lane * 16, 0, 1);

    for (int i = tid; i < 48 * 64; i += 256) {
        int r = i >> 6, c4 = i & 63;
        f32x4 f = {0.f, 0.f, 0.f, 0.f};
        if (r < KNN)
            f = *(const f32x4*)(rpe + ((size_t)n * KNN + r) * 256 + c4 * 4);
        bf16x4 b;
        b[0] = (bf16)f[0]; b[1] = (bf16)f[1];
        b[2] = (bf16)f[2]; b[3] = (bf16)f[3];
        *(bf16x4*)(s_rpe + r * RPE_STR + c4 * 4) = b;
    }

    wait_asynccnt0();      // async gather complete (this wave's issues)
    __syncthreads();       // all waves' LDS writes visible

    // -------- phase 1: p(48x256) = rpe_tile @ Wp via WMMA ------------------
    // wave w owns N-tiles {2w, 2w+1}; 3 M-tiles x 8 K-steps -> 48 WMMAs/wave.
    {
        const int mrow  = lane & 15;
        const int khalf = (lane >> 4) * 8;
        v8f acc[3][2] = {};
        for (int ks = 0; ks < 256; ks += 32) {
            v16bf bfr[2];
            #pragma unroll
            for (int j = 0; j < 2; ++j) {
                int col = (wave * 2 + j) * 16 + mrow;
                const bf16* bb = Wpt + (size_t)col * 256 + ks + khalf;  // L2-hot
                bfr[j] = load_frag(bb, bb + 16);
            }
            #pragma unroll
            for (int mt = 0; mt < 3; ++mt) {
                const bf16* ab = s_rpe + (mt * 16 + mrow) * RPE_STR + ks + khalf;
                v16bf a = load_frag(ab, ab + 16);
                acc[mt][0] = wmma_bf16(a, bfr[0], acc[mt][0]);
                acc[mt][1] = wmma_bf16(a, bfr[1], acc[mt][1]);
            }
        }
        #pragma unroll
        for (int mt = 0; mt < 3; ++mt)
            #pragma unroll
            for (int j = 0; j < 2; ++j) {
                int col = (wave * 2 + j) * 16 + mrow;
                #pragma unroll
                for (int r = 0; r < 8; ++r) {
                    int m = mt * 16 + r + (lane >> 4) * 8;
                    s_p[m * P_STR + col] = acc[mt][j][r];
                }
            }
    }
    __syncthreads();

    // -------- phase 2: scores[h][kk] = (q.k_nb + q1.p) / sqrt(D) -----------
    const float scale = 0.17677669529663687f;   // 1/sqrt(32)
    for (int i = tid; i < NHEAD * KNN; i += 256) {
        int h = i / KNN, kk = i % KNN;
        const float* qh  = s_q  + h * HD;
        const float* q1h = s_q1 + h * HD;
        const float* kr  = s_knb + kk * KV_STR + h * HD;
        const float* pr  = s_p   + kk * P_STR  + h * HD;
        float s = 0.f;
        #pragma unroll
        for (int d = 0; d < HD; ++d) s += qh[d] * kr[d] + q1h[d] * pr[d];
        s_sc[h * 64 + kk] = s * scale;
    }
    __syncthreads();

    // -------- phase 3: softmax over kk (one thread per head; 35 elems) -----
    if (tid < NHEAD) {
        float* row = s_sc + tid * 64;
        float mx = -3.4e38f;
        for (int kk = 0; kk < KNN; ++kk) mx = fmaxf(mx, row[kk]);
        float sum = 0.f;
        for (int kk = 0; kk < KNN; ++kk) {
            float e = __expf(row[kk] - mx);
            row[kk] = e; sum += e;
        }
        float inv = 1.f / sum;
        for (int kk = 0; kk < KNN; ++kk) {
            float pv = row[kk] * inv;
            row[kk] = pv;
            out_p[((size_t)n * NHEAD + tid) * KNN + kk] = pv;
        }
    }
    __syncthreads();

    // -------- phase 4: hidden[c] = sum_kk probs[h][kk] * v_nb[kk][c] -------
    {
        int h = tid >> 5;   // c = tid, head = c/32
        float a = 0.f;
        #pragma unroll 5
        for (int kk = 0; kk < KNN; ++kk)
            a += s_sc[h * 64 + kk] * s_vnb[kk * KV_STR + tid];
        out_h[(size_t)n * 256 + tid] = a;
    }
}

// ---------------------------------------------------------------------------
// Host launcher
// ---------------------------------------------------------------------------
extern "C" void kernel_launch(void* const* d_in, const int* in_sizes, int n_in,
                              void* d_out, int out_size, void* d_ws, size_t ws_size,
                              hipStream_t stream) {
    (void)in_sizes; (void)n_in; (void)out_size; (void)ws_size;

    const float* input_q = (const float*)d_in[0];
    const float* input_k = (const float*)d_in[1];
    const float* input_v = (const float*)d_in[2];
    const float* rpe     = (const float*)d_in[3];
    const int*   knn_idx = (const int*)  d_in[4];
    const float* Wq  = (const float*)d_in[5];
    const float* bq  = (const float*)d_in[6];
    const float* Wq1 = (const float*)d_in[7];
    const float* bq1 = (const float*)d_in[8];
    const float* Wk  = (const float*)d_in[9];
    const float* bk  = (const float*)d_in[10];
    const float* Wv  = (const float*)d_in[11];
    const float* bv  = (const float*)d_in[12];
    const float* Wp  = (const float*)d_in[13];
    // d_in[14] = bp: constant over softmax axis -> cancels exactly; unused.

    // Workspace layout
    const size_t WELEMS = 256 * 256;
    bf16* Wt_q  = (bf16*)d_ws;
    bf16* Wt_q1 = Wt_q  + WELEMS;
    bf16* Wt_k  = Wt_q1 + WELEMS;
    bf16* Wt_v  = Wt_k  + WELEMS;
    bf16* Wt_p  = Wt_v  + WELEMS;
    float* fbuf = (float*)((char*)d_ws + 5 * WELEMS * sizeof(bf16));
    const size_t PELEMS = (size_t)NPTS * CDIM;          // 4,194,304
    float* q_buf  = fbuf;
    float* q1_buf = fbuf + PELEMS;
    float* k_buf  = fbuf + 2 * PELEMS;
    float* v_buf  = fbuf + 3 * PELEMS;

    float* out_hidden = (float*)d_out;                  // [N, 256]
    float* out_probs  = out_hidden + PELEMS;            // [N, 8, 35]

    // 1) weight transpose/convert (65536 elems each -> 256 blocks x 256 thr)
    wt_convert<<<256, 256, 0, stream>>>(Wq,  Wt_q);
    wt_convert<<<256, 256, 0, stream>>>(Wq1, Wt_q1);
    wt_convert<<<256, 256, 0, stream>>>(Wk,  Wt_k);
    wt_convert<<<256, 256, 0, stream>>>(Wv,  Wt_v);
    wt_convert<<<256, 256, 0, stream>>>(Wp,  Wt_p);

    // 2) projections: grid (16384/128, 256/64)
    dim3 ggrid(NPTS / 128, 4);
    gemm_proj<<<ggrid, 256, 0, stream>>>(input_q, Wt_q,  bq,  q_buf);
    gemm_proj<<<ggrid, 256, 0, stream>>>(input_q, Wt_q1, bq1, q1_buf);
    gemm_proj<<<ggrid, 256, 0, stream>>>(input_k, Wt_k,  bk,  k_buf);
    gemm_proj<<<ggrid, 256, 0, stream>>>(input_v, Wt_v,  bv,  v_buf);

    // 3) fused attention: one block per query, ~150 KB dynamic LDS
    rpe_attn<<<NPTS, 256, SMEM_BYTES, stream>>>(rpe, knn_idx, Wt_p,
                                                q_buf, q1_buf, k_buf, v_buf,
                                                out_hidden, out_probs);
}
